// SpMiddlePillarEncoderVgg_37486474559968
// MI455X (gfx1250) — compile-verified, matrix operand order checked by reading, not run on hardware
//
#include <hip/hip_runtime.h>
#include <hip/hip_bf16.h>

// ---------------------------------------------------------------------------
// SpMiddlePillarEncoderVgg for MI455X (gfx1250, wave32, WMMA).
//
// Inputs (setup_inputs dict order, nested dicts flattened in insertion order):
//   d_in[0] xyz            (240000*3)  f32
//   d_in[1] xyz_batch_cnt  (2)         i32
//   d_in[2] pt_features    (240000*32) f32
//   d_in[3 + 5*L + 0] = layer L conv weight (3*3*cin*cout, HWIO) f32
//   d_in[3 + 5*L + 1..4] = gamma, beta, mean, var (cout) f32
// Output: x1|x2|x3|x4|x5 concatenated, NHWC f32 (31,981,568 floats).
//
// Conv = implicit GEMM on v_wmma_f32_16x16x32_f16: each wave owns a
// 16-pixel x 32-outch tile (two accumulators sharing one A fragment),
// 4 waves/block share per-tap B fragments staged in LDS (2 barriers / tap,
// up to 16 WMMAs between barriers). Activations ping-pong through two 64MiB
// buffers that stay resident in the 192MB L2.
// ---------------------------------------------------------------------------

typedef __attribute__((ext_vector_type(16))) _Float16 v16h;
typedef __attribute__((ext_vector_type(8)))  float    v8f;

#define GRID_H 512
#define GRID_W 512
#define NBATCH 2
#define C0     32
#define NPTS   240000

// --------------------------- pillar pooling --------------------------------

__device__ __forceinline__ unsigned enc_key(float f) {
  unsigned u = __float_as_uint(f);
  return (u & 0x80000000u) ? ~u : (u | 0x80000000u);
}
__device__ __forceinline__ float dec_key(unsigned k) {
  unsigned u = (k & 0x80000000u) ? (k ^ 0x80000000u) : ~k;
  return __uint_as_float(u);
}

__global__ void pillar_scatter(const float* __restrict__ xyz,
                               const int* __restrict__ bcnt,
                               const float* __restrict__ feat,
                               unsigned* __restrict__ keys,
                               float* __restrict__ mask1) {
  int n = blockIdx.x * blockDim.x + threadIdx.x;
  if (n >= NPTS) return;
  int bid = (n < bcnt[0]) ? 0 : 1;
  float x = xyz[n * 3 + 0];
  float y = xyz[n * 3 + 1];
  int ix = (int)floorf((x + 51.2f) * 5.0f);
  int iy = (int)floorf((y + 51.2f) * 5.0f);
  ix = min(max(ix, 0), GRID_W - 1);
  iy = min(max(iy, 0), GRID_H - 1);
  long seg = ((long)bid * GRID_H + iy) * GRID_W + ix;
  mask1[seg] = 1.0f;                        // benign race: all write 1.0
  unsigned* kp = keys + seg * C0;
  const float* fp = feat + (long)n * C0;
#pragma unroll
  for (int c = 0; c < C0; ++c) atomicMax(kp + c, enc_key(fp[c]));
}

__global__ void pillar_decode(unsigned* __restrict__ keys,  // in-place -> f32
                              const float* __restrict__ mask1) {
  long i = (long)blockIdx.x * blockDim.x + threadIdx.x;
  const long total = (long)NBATCH * GRID_H * GRID_W * C0;
  if (i >= total) return;
  unsigned k = keys[i];
  float v = (mask1[i >> 5] > 0.0f) ? dec_key(k) : 0.0f;
  ((float*)keys)[i] = v;
}

// --------------------------- mask max-pool ---------------------------------
// 3x3 window, stride 2, pad 1, init 0 (matches reduce_window in reference).

__global__ void pool_mask(const float* __restrict__ mi, float* __restrict__ mo,
                          int Hi, int Wi, int Ho, int Wo) {
  int i = blockIdx.x * blockDim.x + threadIdx.x;
  int total = NBATCH * Ho * Wo;
  if (i >= total) return;
  int x = i % Wo, y = (i / Wo) % Ho, b = i / (Wo * Ho);
  float m = 0.0f;
#pragma unroll
  for (int dy = -1; dy <= 1; ++dy)
#pragma unroll
    for (int dx = -1; dx <= 1; ++dx) {
      int yy = 2 * y + dy, xx = 2 * x + dx;
      if (yy >= 0 && yy < Hi && xx >= 0 && xx < Wi)
        m = fmaxf(m, mi[((long)b * Hi + yy) * Wi + xx]);
    }
  mo[i] = m;
}

// --------------------------- weight packing --------------------------------
// Pack HWIO f32 weights into f16 WMMA B-fragments (16x16x32 f16 layout,
// ISA 7.12.2): fragment f = ((tap*KB + kb)*NB + nb), 32 lanes x 16 halfs.
// lane = Ncol + 16*kg ; half h<8 -> K = kg*8+h ; h>=8 -> K = 16+kg*8+(h-8).

__global__ void pack_weights(const float* __restrict__ w,
                             _Float16* __restrict__ dst,
                             int cin, int cout, int total) {
  int idx = blockIdx.x * blockDim.x + threadIdx.x;
  if (idx >= total) return;
  int h    = idx & 15;
  int lane = (idx >> 4) & 31;
  int f    = idx >> 9;
  int NB = cout >> 4, KB = cin >> 5;
  int nb = f % NB;
  int r  = f / NB;
  int kb = r % KB;
  int t  = r / KB;
  int kg = lane >> 4;
  int n  = (lane & 15) + nb * 16;
  int K  = (h < 8) ? (kg * 8 + h) : (16 + kg * 8 + (h - 8));
  int k  = kb * 32 + K;
  dst[idx] = (_Float16)w[((long)t * cin + k) * cout + n];
}

// --------------------------- WMMA conv3x3 ----------------------------------
// Implicit GEMM: D[16 pixels x 32 outch] += A[16 x 32] * B[32 x 32] over
// K = 9 taps * cin. Per tap, all KB*2 B fragments are staged in LDS once
// (2 barriers per tap), then KB A-gathers each feed TWO wmma ops.

#define WAVES_PER_BLOCK 4
#define MAX_KB 8                      // cin <= 256
// LDS: 2 ntiles * MAX_KB fragments * 512 halfs = 16 KiB

__global__ __launch_bounds__(128) void conv3x3_wmma(
    const float* __restrict__ in, float* __restrict__ out,
    const _Float16* __restrict__ wpack,
    const float* __restrict__ gamma, const float* __restrict__ beta,
    const float* __restrict__ mean, const float* __restrict__ var,
    const float* __restrict__ mask,   // at output resolution, may be null
    int Hin, int Win, int Ho, int Wo, int cin, int cout, int stride,
    int relu) {
  const int lane  = threadIdx.x & 31;
  const int wave  = threadIdx.x >> 5;
  const int nt0   = blockIdx.y * 2;          // first of two outch tiles
  const int mtile = blockIdx.x * WAVES_PER_BLOCK + wave;
  const int KB = cin >> 5;
  const int NB = cout >> 4;

  const int l16 = lane & 15;   // A row index m / D column index n
  const int kg  = lane >> 4;   // K group (0: K 0-7,16-23 ; 1: K 8-15,24-31)

  // pixel handled by this lane's A row
  const int p  = mtile * 16 + l16;
  const int ox = p % Wo;
  const int oy = (p / Wo) % Ho;
  const int b  = p / (Wo * Ho);

  __shared__ alignas(32) _Float16 sB[2 * MAX_KB * 512];

  union { v8f v; float f[8]; } acc0, acc1;
#pragma unroll
  for (int r = 0; r < 8; ++r) { acc0.f[r] = 0.0f; acc1.f[r] = 0.0f; }

#pragma unroll 1
  for (int t = 0; t < 9; ++t) {
    const int dy = t / 3 - 1, dx = t % 3 - 1;
    const int iy = oy * stride + dy;
    const int ix = ox * stride + dx;
    const bool inb = (iy >= 0) & (iy < Hin) & (ix >= 0) & (ix < Win);
    const float* src = in + (((long)b * Hin + iy) * Win + ix) * cin;
    if (inb) __builtin_prefetch(src, 0, 0);   // global_prefetch_b8

    // ---- stage this tap's B fragments (both ntiles, all kb) into LDS ----
    __syncthreads();
    {
      const int nchunk = 2 * KB * 64;        // 16B chunks
      for (int c = threadIdx.x; c < nchunk; c += 128) {
        const int fi = c >> 6;               // fragment = kb*2 + which
        const int wi = c & 63;               // 16B chunk within fragment
        const int kb = fi >> 1;
        const int which = fi & 1;
        const uint4* s = (const uint4*)(wpack +
            ((long)((t * KB + kb) * NB + nt0 + which) << 9) + (wi << 3));
        ((uint4*)sB)[(fi << 6) + wi] = *s;
      }
    }
    __syncthreads();

#pragma unroll 1
    for (int kb = 0; kb < KB; ++kb) {
      // ---- gather A fragment: two runs of 8 consecutive channels ----
      union { v16h v; _Float16 h[16]; } a;
      if (inb) {
        const float4* s0 = (const float4*)(src + kb * 32 + kg * 8);
        const float4* s1 = (const float4*)(src + kb * 32 + 16 + kg * 8);
        float4 r0 = s0[0], r1 = s0[1], r2 = s1[0], r3 = s1[1];
        a.h[0]  = (_Float16)r0.x; a.h[1]  = (_Float16)r0.y;
        a.h[2]  = (_Float16)r0.z; a.h[3]  = (_Float16)r0.w;
        a.h[4]  = (_Float16)r1.x; a.h[5]  = (_Float16)r1.y;
        a.h[6]  = (_Float16)r1.z; a.h[7]  = (_Float16)r1.w;
        a.h[8]  = (_Float16)r2.x; a.h[9]  = (_Float16)r2.y;
        a.h[10] = (_Float16)r2.z; a.h[11] = (_Float16)r2.w;
        a.h[12] = (_Float16)r3.x; a.h[13] = (_Float16)r3.y;
        a.h[14] = (_Float16)r3.z; a.h[15] = (_Float16)r3.w;
      } else {
#pragma unroll
        for (int i = 0; i < 16; ++i) a.h[i] = (_Float16)0.0f;
      }

      v16h b0 = *reinterpret_cast<const v16h*>(&sB[((kb * 2 + 0) << 9) + (lane << 4)]);
      v16h b1 = *reinterpret_cast<const v16h*>(&sB[((kb * 2 + 1) << 9) + (lane << 4)]);

      acc0.v = __builtin_amdgcn_wmma_f32_16x16x32_f16(
          false, a.v, false, b0, (short)0, acc0.v, false, false);
      acc1.v = __builtin_amdgcn_wmma_f32_16x16x32_f16(
          false, a.v, false, b1, (short)0, acc1.v, false, false);
    }
  }

  // ---- epilogue: BN (scale/shift) + optional ReLU + sparsity mask ----
  const int ch0 = nt0 * 16 + l16;
  const int ch1 = ch0 + 16;
  const float sc0 = gamma[ch0] * rsqrtf(var[ch0] + 1e-3f);
  const float sh0 = beta[ch0] - mean[ch0] * sc0;
  const float sc1 = gamma[ch1] * rsqrtf(var[ch1] + 1e-3f);
  const float sh1 = beta[ch1] - mean[ch1] * sc1;
#pragma unroll
  for (int r = 0; r < 8; ++r) {
    const int m = r + (kg << 3);         // D row: lanes16-31 hold M=8..15
    const long q = (long)mtile * 16 + m; // flat output pixel
    float mk = mask ? mask[q] : 1.0f;
    float v0 = acc0.f[r] * sc0 + sh0;
    float v1 = acc1.f[r] * sc1 + sh1;
    if (relu) { v0 = fmaxf(v0, 0.0f); v1 = fmaxf(v1, 0.0f); }
    out[q * cout + ch0] = v0 * mk;
    out[q * cout + ch1] = v1 * mk;
  }
}

// --------------------------- host-side driver ------------------------------

extern "C" void kernel_launch(void* const* d_in, const int* in_sizes, int n_in,
                              void* d_out, int out_size, void* d_ws,
                              size_t ws_size, hipStream_t stream) {
  (void)in_sizes; (void)n_in; (void)out_size; (void)ws_size;

  const float* xyz  = (const float*)d_in[0];
  const int*   bcnt = (const int*)d_in[1];
  const float* feat = (const float*)d_in[2];

  static const int CIN[14]  = {32, 32, 32, 64, 64, 64, 128, 128,
                               128, 256, 256, 256, 256, 256};
  static const int COUT[14] = {32, 32, 64, 64, 64, 128, 128, 128,
                               256, 256, 256, 256, 256, 256};
  static const int HIN[14]  = {512, 512, 512, 256, 256, 256, 128, 128,
                               128, 64, 64, 64, 32, 32};
  static const int HOUT[14] = {512, 512, 256, 256, 256, 128, 128, 128,
                               64, 64, 64, 32, 32, 32};
  static const int STR[14]  = {1, 1, 2, 1, 1, 2, 1, 1, 2, 1, 1, 2, 1, 1};
  static const int RELU[14] = {0, 1, 1, 1, 1, 1, 1, 1, 1, 1, 1, 1, 1, 1};

  // workspace carve-up
  char* ws = (char*)d_ws;
  const size_t PING = 64u << 20;               // 67,108,864 B each
  float* wsA = (float*)ws;
  float* wsB = (float*)(ws + PING);
  float* m1  = (float*)(ws + 2 * PING);
  float* m2  = m1 + (size_t)NBATCH * 512 * 512;
  float* m3  = m2 + (size_t)NBATCH * 256 * 256;
  float* m4  = m3 + (size_t)NBATCH * 128 * 128;
  _Float16* wp = (_Float16*)(m4 + (size_t)NBATCH * 64 * 64);

  // output carve-up
  float* x1 = (float*)d_out;
  float* x2 = x1 + (size_t)NBATCH * 512 * 512 * 32;
  float* x3 = x2 + (size_t)NBATCH * 256 * 256 * 64;
  float* x4 = x3 + (size_t)NBATCH * 128 * 128 * 128;
  float* x5 = x4 + (size_t)NBATCH * 64 * 64 * 256;

  // 1) pillar pooling (segment-max via monotone-key atomics)
  hipMemsetAsync(wsA, 0, (size_t)NBATCH * 512 * 512 * 32 * 4, stream);
  hipMemsetAsync(m1, 0, (size_t)NBATCH * 512 * 512 * 4, stream);
  pillar_scatter<<<(NPTS + 255) / 256, 256, 0, stream>>>(
      xyz, bcnt, feat, (unsigned*)wsA, m1);
  {
    long total = (long)NBATCH * 512 * 512 * 32;
    pillar_decode<<<(unsigned)((total + 255) / 256), 256, 0, stream>>>(
        (unsigned*)wsA, m1);
  }

  // 2) pack all weights to f16 B-fragments
  size_t woff[15];
  woff[0] = 0;
  for (int l = 0; l < 14; ++l) woff[l + 1] = woff[l] + (size_t)9 * CIN[l] * COUT[l];
  for (int l = 0; l < 14; ++l) {
    int total = 9 * CIN[l] * COUT[l];
    pack_weights<<<(total + 255) / 256, 256, 0, stream>>>(
        (const float*)d_in[3 + 5 * l], wp + woff[l], CIN[l], COUT[l], total);
  }

  // 3) downsampled sparsity masks
  pool_mask<<<(NBATCH * 256 * 256 + 255) / 256, 256, 0, stream>>>(m1, m2, 512, 512, 256, 256);
  pool_mask<<<(NBATCH * 128 * 128 + 255) / 256, 256, 0, stream>>>(m2, m3, 256, 256, 128, 128);
  pool_mask<<<(NBATCH * 64 * 64 + 255) / 256, 256, 0, stream>>>(m3, m4, 128, 128, 64, 64);

  // 4) conv chain
  const float* lin[14] = {wsA, wsB, x1, wsA, wsB, x2, wsA, wsB,
                          x3, wsA, wsB, x4, wsA, wsB};
  float* lout[14] = {wsB, x1, wsA, wsB, x2, wsA, wsB, x3,
                     wsA, wsB, x4, wsA, wsB, x5};
  const float* lmask[14] = {m1, m1, m2, m2, m2, m3, m3, m3,
                            m4, m4, m4, nullptr, nullptr, nullptr};

  for (int l = 0; l < 14; ++l) {
    const int Ho = HOUT[l];
    const int Mtiles = (NBATCH * Ho * Ho) / 16;   // divisible by 4 for all layers
    dim3 grid(Mtiles / WAVES_PER_BLOCK, COUT[l] / 32);
    conv3x3_wmma<<<grid, 128, 0, stream>>>(
        lin[l], lout[l], wp + woff[l],
        (const float*)d_in[3 + 5 * l + 1], (const float*)d_in[3 + 5 * l + 2],
        (const float*)d_in[3 + 5 * l + 3], (const float*)d_in[3 + 5 * l + 4],
        lmask[l], HIN[l], HIN[l], Ho, Ho, CIN[l], COUT[l], STR[l], RELU[l]);
  }
}
